// W8A8B32O32LinearWithoutScaling_32968168964164
// MI455X (gfx1250) — 8-bit matrix operand order repaired and checked by reading, compiled, not run on hardware
//
#include <hip/hip_runtime.h>
#include <stdint.h>

typedef __attribute__((ext_vector_type(8))) int v8i;
typedef __attribute__((ext_vector_type(4))) int v4i;
typedef __attribute__((ext_vector_type(2))) int v2i;

#define GAS __attribute__((address_space(1)))

constexpr int IN_DIM  = 4096;
constexpr int OUT_DIM = 4096;

union AReg { v8i v8; v2i v2[4]; };
union BReg { v8i v8; v4i v4[2]; };

__global__ __launch_bounds__(256, 1)
void w8a8b32o32_gemm_kernel(const int8_t* __restrict__ x_,
                            const int8_t* __restrict__ w_,
                            const int*    __restrict__ bias_,
                            int*          __restrict__ out_)
{
    const GAS int8_t* x    = (const GAS int8_t*)x_;
    const GAS int8_t* w    = (const GAS int8_t*)w_;
    const GAS int*    bias = (const GAS int*)bias_;
    GAS int*          out  = (GAS int*)out_;

    const int lane = threadIdx.x & 31;
    const int wave = threadIdx.x >> 5;
    const int half = lane >> 4;
    const int l16  = lane & 15;

    const int waveM = wave & 1;
    const int waveN = wave >> 1;

    const int m0 = blockIdx.y * 128 + waveM * 64;
    const int n0 = blockIdx.x * 256 + waveN * 64;

    v8i acc[4][4];
#pragma unroll
    for (int ni = 0; ni < 4; ++ni) {
        const int bv = bias[n0 + ni * 16 + l16];
#pragma unroll
        for (int mi = 0; mi < 4; ++mi) {
#pragma unroll
            for (int r = 0; r < 8; ++r) acc[mi][ni][r] = bv;
        }
    }

    const GAS int8_t* aPtr[4];
#pragma unroll
    for (int mi = 0; mi < 4; ++mi)
        aPtr[mi] = x + (size_t)(m0 + mi * 16 + l16) * IN_DIM + half * 16;

    const GAS int8_t* bPtr[4];
#pragma unroll
    for (int ni = 0; ni < 4; ++ni)
        bPtr[ni] = w + (size_t)(n0 + ni * 16 + l16) * IN_DIM + half * 16;

    for (int k0 = 0; k0 < IN_DIM; k0 += 64) {
        BReg b[4];
#pragma unroll
        for (int ni = 0; ni < 4; ++ni) {
            b[ni].v4[0] = *(const GAS v4i*)(bPtr[ni] + 0);
            b[ni].v4[1] = *(const GAS v4i*)(bPtr[ni] + 32);
            bPtr[ni] += 64;
        }
        AReg a[4];
#pragma unroll
        for (int mi = 0; mi < 4; ++mi) {
            a[mi].v2[0] = *(const GAS v2i*)(aPtr[mi] + 0);
            a[mi].v2[1] = *(const GAS v2i*)(aPtr[mi] + 8);
            a[mi].v2[2] = *(const GAS v2i*)(aPtr[mi] + 32);
            a[mi].v2[3] = *(const GAS v2i*)(aPtr[mi] + 40);
            aPtr[mi] += 64;
        }
#pragma unroll
        for (int mi = 0; mi < 4; ++mi) {
#pragma unroll
            for (int ni = 0; ni < 4; ++ni) {
                acc[mi][ni] = __builtin_amdgcn_wmma_i32_16x16x64_iu8(
                    true, a[mi].v8, true, b[ni].v8, acc[mi][ni], false, false);
            }
        }
    }

#pragma unroll
    for (int mi = 0; mi < 4; ++mi) {
#pragma unroll
        for (int ni = 0; ni < 4; ++ni) {
            const int col  = n0 + ni * 16 + l16;
            const int rowb = m0 + mi * 16 + half * 8;
#pragma unroll
            for (int r = 0; r < 8; ++r) {
                __builtin_nontemporal_store(acc[mi][ni][r],
                                            &out[(size_t)(rowb + r) * OUT_DIM + col]);
            }
        }
    }
}

extern "C" void kernel_launch(void* const* d_in, const int* in_sizes, int n_in,
                              void* d_out, int out_size, void* d_ws, size_t ws_size,
                              hipStream_t stream) {
    const int8_t* x    = (const int8_t*)d_in[0];
    const int8_t* w    = (const int8_t*)d_in[1];
    const int*    bias = (const int*)d_in[2];
    int*          out  = (int*)d_out;

    const int tokens = in_sizes[0] / IN_DIM;
    dim3 grid(OUT_DIM / 256, tokens / 128);
    dim3 block(256);
    hipLaunchKernelGGL(w8a8b32o32_gemm_kernel, grid, block, 0, stream,
                       x, w, bias, out);
}
